// PerformerAttentionLayer_83683142795477
// MI455X (gfx1250) — matrix operand order repaired and checked by reading, not compiled, not run on hardware
//
#include <hip/hip_runtime.h>
#include <math.h>

// ---------------------------------------------------------------------------
// Performer attention layer for MI455X (gfx1250), wave32 + WMMA.
// Linear-attention reassociation removes the 8192x8192 QK matrix.
// f32 WMMA (16x16x4) pre-exp (async global->LDS for A tiles),
// bf16 WMMA (16x16x32) post-exp (staged register loads + on-the-fly cvt).
// ---------------------------------------------------------------------------

#define NROWS 8192
#define DIM   256

typedef __bf16 bf16;
typedef __attribute__((ext_vector_type(16))) __bf16 v16bf;
typedef __attribute__((ext_vector_type(8)))  __bf16 v8bf;
typedef __attribute__((ext_vector_type(8)))  float  v8f;
typedef __attribute__((ext_vector_type(2)))  float  v2f;

union frag16 { v16bf v; v8bf h[2]; };
union pk4    { bf16 b[4]; unsigned long long u; };

__device__ __forceinline__ bf16 f2bf(float f) {
    union { float f; unsigned int u; } in; in.f = f;
    unsigned int u = in.u;
    u += 0x7FFFu + ((u >> 16) & 1u);               // round-to-nearest-even
    union { unsigned short s; bf16 b; } out;
    out.s = (unsigned short)(u >> 16);
    return out.b;
}

__device__ __forceinline__ bool is_inf_f32(float f) {
    union { float f; unsigned int u; } c; c.f = f;
    return (c.u & 0x7FFFFFFFu) == 0x7F800000u;     // +inf or -inf
}

// Async global -> LDS copy of 16 bytes per lane (CDNA5 TDM-lite path).
// lds_ptr low 32 bits are the LDS byte offset (flat aperture truncation).
__device__ __forceinline__ void async_copy_b128(void* lds_ptr, const void* gptr) {
    unsigned lds_off = (unsigned)(unsigned long long)lds_ptr;
    unsigned long long ga = (unsigned long long)gptr;
    asm volatile("global_load_async_to_lds_b128 %0, %1, off"
                 :: "v"(lds_off), "v"(ga) : "memory");
}

__device__ __forceinline__ void wait_async() {
    asm volatile("s_wait_asynccnt 0" ::: "memory");
}

// ---------------------------------------------------------------------------
// bf16 GEMM: C[M,N] = A[M,K] @ B[K,N] (+bias)(+relu). 128x128 tile, BK=32.
// mode: 0 = plain, 1 = +bias, 2 = +bias then relu
// ---------------------------------------------------------------------------
__global__ __launch_bounds__(256)
void gemm_bf16_kernel(const float* __restrict__ A, const float* __restrict__ B,
                      const float* __restrict__ bias, float* __restrict__ C,
                      int M, int N, int K, int mode)
{
    __shared__ __align__(16) bf16 As[128 * 40];   // As[m*40 + k]
    __shared__ __align__(16) bf16 Bs[128 * 40];   // Bs[n*40 + k] (transposed)
    const int t    = threadIdx.x;
    const int lane = t & 31;
    const int wid  = t >> 5;
    const int wm   = wid & 1;      // wave M index (0..1)  -> 64 rows
    const int wn   = wid >> 1;     // wave N index (0..3)  -> 32 cols
    const int m0   = blockIdx.y * 128;
    const int n0   = blockIdx.x * 128;
    const int hi   = lane >> 4;
    const int l16  = lane & 15;

    v8f acc[4][2] = {};

    for (int k0 = 0; k0 < K; k0 += 32) {
        // Phase 1: issue all 8 tile loads into registers (stay in flight).
        float4 ra[4], rb[4];
        #pragma unroll
        for (int i = 0; i < 4; ++i) {
            int flat = t + 256 * i;
            int row  = flat >> 3;
            int kq   = (flat & 7) << 2;
            ra[i] = *(const float4*)(A + (m0 + row) * K + k0 + kq);
        }
        #pragma unroll
        for (int i = 0; i < 4; ++i) {
            int flat = t + 256 * i;
            int kk   = flat >> 5;
            int nq   = (flat & 31) << 2;
            rb[i] = *(const float4*)(B + (k0 + kk) * N + n0 + nq);
        }
        if (k0 + 32 < K) {
            __builtin_prefetch(A + (m0 + (t >> 1)) * K + k0 + 32, 0, 1);
            __builtin_prefetch(B + (k0 + 32 + (t >> 3)) * N + n0, 0, 1);
        }
        // Phase 2: convert + store to LDS.
        #pragma unroll
        for (int i = 0; i < 4; ++i) {
            int flat = t + 256 * i;
            int row  = flat >> 3;
            int kq   = (flat & 7) << 2;
            pk4 p;
            p.b[0] = f2bf(ra[i].x); p.b[1] = f2bf(ra[i].y);
            p.b[2] = f2bf(ra[i].z); p.b[3] = f2bf(ra[i].w);
            *(unsigned long long*)(As + row * 40 + kq) = p.u;
        }
        #pragma unroll
        for (int i = 0; i < 4; ++i) {
            int flat = t + 256 * i;
            int kk   = flat >> 5;
            int nq   = (flat & 31) << 2;
            Bs[(nq + 0) * 40 + kk] = f2bf(rb[i].x);
            Bs[(nq + 1) * 40 + kk] = f2bf(rb[i].y);
            Bs[(nq + 2) * 40 + kk] = f2bf(rb[i].z);
            Bs[(nq + 3) * 40 + kk] = f2bf(rb[i].w);
        }
        __syncthreads();

        frag16 af[4], bf_[2];
        #pragma unroll
        for (int r = 0; r < 4; ++r) {     // A frag: lane holds row m, K 8hi..+8 and 16+8hi..+8
            int m = wm * 64 + r * 16 + l16;
            af[r].h[0] = *(const v8bf*)(As + m * 40 + 8 * hi);
            af[r].h[1] = *(const v8bf*)(As + m * 40 + 16 + 8 * hi);
        }
        #pragma unroll
        for (int c = 0; c < 2; ++c) {     // B frag: lane holds col n, K 8hi..+8 and 16+8hi..+8
            int n = wn * 32 + c * 16 + l16;
            bf_[c].h[0] = *(const v8bf*)(Bs + n * 40 + 8 * hi);
            bf_[c].h[1] = *(const v8bf*)(Bs + n * 40 + 16 + 8 * hi);
        }
        #pragma unroll
        for (int r = 0; r < 4; ++r)
            #pragma unroll
            for (int c = 0; c < 2; ++c)
                acc[r][c] = __builtin_amdgcn_wmma_f32_16x16x32_bf16(
                    false, af[r].v, false, bf_[c].v, (short)0, acc[r][c], false, false);
        __syncthreads();
    }

    #pragma unroll
    for (int c = 0; c < 2; ++c) {
        int col  = n0 + wn * 32 + c * 16 + l16;
        float bv = (mode != 0 && bias != nullptr) ? bias[col] : 0.0f;
        #pragma unroll
        for (int r = 0; r < 4; ++r) {
            #pragma unroll
            for (int e = 0; e < 8; ++e) {
                int row   = m0 + wm * 64 + r * 16 + hi * 8 + e;
                float val = acc[r][c][e] + bv;
                if (mode == 2) val = fmaxf(val, 0.0f);
                C[row * N + col] = val;
            }
        }
    }
}

// ---------------------------------------------------------------------------
// f32 GEMM (exact): C = A @ B using V_WMMA_F32_16X16X4_F32, BK=32.
// A tile copied global->LDS with CDNA5 async-load (no VGPR roundtrip);
// B tile staged through registers (needs transpose into LDS).
// ---------------------------------------------------------------------------
__global__ __launch_bounds__(256)
void gemm_f32_kernel(const float* __restrict__ A, const float* __restrict__ B,
                     float* __restrict__ C, int M, int N, int K)
{
    __shared__ __align__(16) float As[128 * 36];  // As[m*36 + k]
    __shared__ __align__(16) float Bs[128 * 36];  // Bs[n*36 + k] (transposed)
    const int t    = threadIdx.x;
    const int lane = t & 31;
    const int wid  = t >> 5;
    const int wm   = wid & 1;
    const int wn   = wid >> 1;
    const int m0   = blockIdx.y * 128;
    const int n0   = blockIdx.x * 128;
    const int hi   = lane >> 4;
    const int l16  = lane & 15;

    v8f acc[4][2] = {};

    for (int k0 = 0; k0 < K; k0 += 32) {
        // A tile: async DMA, 16B per lane per issue (4 issues -> 16KB tile).
        #pragma unroll
        for (int i = 0; i < 4; ++i) {
            int flat = t + 256 * i;
            int row  = flat >> 3;
            int kq   = (flat & 7) << 2;
            async_copy_b128(As + row * 36 + kq,
                            A + (m0 + row) * K + k0 + kq);
        }
        // B tile: stage in registers, then transpose-store.
        float4 rb[4];
        #pragma unroll
        for (int i = 0; i < 4; ++i) {
            int flat = t + 256 * i;
            int kk   = flat >> 5;
            int nq   = (flat & 31) << 2;
            rb[i] = *(const float4*)(B + (k0 + kk) * N + n0 + nq);
        }
        if (k0 + 32 < K) {
            __builtin_prefetch(A + (m0 + (t >> 1)) * K + k0 + 32, 0, 1);
            __builtin_prefetch(B + (k0 + 32 + (t >> 3)) * N + n0, 0, 1);
        }
        #pragma unroll
        for (int i = 0; i < 4; ++i) {
            int flat = t + 256 * i;
            int kk   = flat >> 5;
            int nq   = (flat & 31) << 2;
            Bs[(nq + 0) * 36 + kk] = rb[i].x;
            Bs[(nq + 1) * 36 + kk] = rb[i].y;
            Bs[(nq + 2) * 36 + kk] = rb[i].z;
            Bs[(nq + 3) * 36 + kk] = rb[i].w;
        }
        wait_async();                 // our async LDS writes have landed
        __syncthreads();

        #pragma unroll
        for (int ks = 0; ks < 32; ks += 4) {
            v2f av[4], bv[2];
            #pragma unroll
            for (int r = 0; r < 4; ++r) {   // A frag: {K=ks+2hi, ks+2hi+1} of row m
                int m = wm * 64 + r * 16 + l16;
                av[r] = *(const v2f*)(As + m * 36 + ks + 2 * hi);
            }
            #pragma unroll
            for (int c = 0; c < 2; ++c) {
                int n = wn * 32 + c * 16 + l16;
                bv[c] = *(const v2f*)(Bs + n * 36 + ks + 2 * hi);
            }
            #pragma unroll
            for (int r = 0; r < 4; ++r)
                #pragma unroll
                for (int c = 0; c < 2; ++c)
                    acc[r][c] = __builtin_amdgcn_wmma_f32_16x16x4_f32(
                        false, av[r], false, bv[c], (short)0, acc[r][c], false, false);
        }
        __syncthreads();
    }

    #pragma unroll
    for (int c = 0; c < 2; ++c) {
        int col = n0 + wn * 32 + c * 16 + l16;
        #pragma unroll
        for (int r = 0; r < 4; ++r)
            #pragma unroll
            for (int e = 0; e < 8; ++e) {
                int row = m0 + wm * 64 + r * 16 + hi * 8 + e;
                C[row * N + col] = acc[r][c][e];
            }
    }
}

// ---------------------------------------------------------------------------
// Split-K bf16 GEMM: Spart[z] = ekp^T @ v over K-chunk of 512.
// A[f,i] = ekp[i,f] (transposed load), B[i,h] = v[i,h]. M=N=256.
// ---------------------------------------------------------------------------
__global__ __launch_bounds__(256)
void splitk_bf16_kernel(const float* __restrict__ Ekp, const float* __restrict__ V,
                        float* __restrict__ Spart)
{
    __shared__ __align__(16) bf16 As[128 * 40];
    __shared__ __align__(16) bf16 Bs[128 * 40];
    const int t    = threadIdx.x;
    const int lane = t & 31;
    const int wid  = t >> 5;
    const int wm   = wid & 1;
    const int wn   = wid >> 1;
    const int m0   = blockIdx.y * 128;          // f
    const int n0   = blockIdx.x * 128;          // h
    const int kb   = blockIdx.z * 512;          // i chunk
    const int hi   = lane >> 4;
    const int l16  = lane & 15;

    v8f acc[4][2] = {};

    for (int k0 = kb; k0 < kb + 512; k0 += 32) {
        float4 ra[4], rb[4];
        #pragma unroll
        for (int i = 0; i < 4; ++i) {           // ekp rows, contiguous along f
            int flat = t + 256 * i;
            int kk   = flat >> 5;
            int fq   = (flat & 31) << 2;
            ra[i] = *(const float4*)(Ekp + (k0 + kk) * DIM + m0 + fq);
        }
        #pragma unroll
        for (int i = 0; i < 4; ++i) {           // v rows, contiguous along h
            int flat = t + 256 * i;
            int kk   = flat >> 5;
            int nq   = (flat & 31) << 2;
            rb[i] = *(const float4*)(V + (k0 + kk) * DIM + n0 + nq);
        }
        #pragma unroll
        for (int i = 0; i < 4; ++i) {           // As[f*40+k] = ekp[(k0+k)*DIM + m0+f]
            int flat = t + 256 * i;
            int kk   = flat >> 5;
            int fq   = (flat & 31) << 2;
            As[(fq + 0) * 40 + kk] = f2bf(ra[i].x);
            As[(fq + 1) * 40 + kk] = f2bf(ra[i].y);
            As[(fq + 2) * 40 + kk] = f2bf(ra[i].z);
            As[(fq + 3) * 40 + kk] = f2bf(ra[i].w);
        }
        #pragma unroll
        for (int i = 0; i < 4; ++i) {           // Bs[h*40+k] = v[(k0+k)*DIM + n0+h]
            int flat = t + 256 * i;
            int kk   = flat >> 5;
            int nq   = (flat & 31) << 2;
            Bs[(nq + 0) * 40 + kk] = f2bf(rb[i].x);
            Bs[(nq + 1) * 40 + kk] = f2bf(rb[i].y);
            Bs[(nq + 2) * 40 + kk] = f2bf(rb[i].z);
            Bs[(nq + 3) * 40 + kk] = f2bf(rb[i].w);
        }
        __syncthreads();

        frag16 af[4], bf_[2];
        #pragma unroll
        for (int r = 0; r < 4; ++r) {
            int m = wm * 64 + r * 16 + l16;
            af[r].h[0] = *(const v8bf*)(As + m * 40 + 8 * hi);
            af[r].h[1] = *(const v8bf*)(As + m * 40 + 16 + 8 * hi);
        }
        #pragma unroll
        for (int c = 0; c < 2; ++c) {
            int n = wn * 32 + c * 16 + l16;
            bf_[c].h[0] = *(const v8bf*)(Bs + n * 40 + 8 * hi);
            bf_[c].h[1] = *(const v8bf*)(Bs + n * 40 + 16 + 8 * hi);
        }
        #pragma unroll
        for (int r = 0; r < 4; ++r)
            #pragma unroll
            for (int c = 0; c < 2; ++c)
                acc[r][c] = __builtin_amdgcn_wmma_f32_16x16x32_bf16(
                    false, af[r].v, false, bf_[c].v, (short)0, acc[r][c], false, false);
        __syncthreads();
    }

    float* outp = Spart + (size_t)blockIdx.z * DIM * DIM;
    #pragma unroll
    for (int c = 0; c < 2; ++c) {
        int col = n0 + wn * 32 + c * 16 + l16;
        #pragma unroll
        for (int r = 0; r < 4; ++r)
            #pragma unroll
            for (int e = 0; e < 8; ++e) {
                int row = m0 + wm * 64 + r * 16 + hi * 8 + e;
                outp[row * DIM + col] = acc[r][c][e];
            }
    }
}

// ---------------------------------------------------------------------------
// Row-wise: clamp inf -> 1e10, subtract row max, exp. One block per row.
// ---------------------------------------------------------------------------
__global__ void rowexp_kernel(const float* __restrict__ in, float* __restrict__ out)
{
    __shared__ float red[256];
    const int row = blockIdx.x, t = threadIdx.x;
    float v = in[row * DIM + t];
    if (is_inf_f32(v)) v = 1.0e10f;
    red[t] = v;
    __syncthreads();
    for (int s = 128; s > 0; s >>= 1) {
        if (t < s) red[t] = fmaxf(red[t], red[t + s]);
        __syncthreads();
    }
    out[row * DIM + t] = expf(v - red[0]);
}

// Column partial sums of ekp: zpart[b][f] = sum over 256-row stripe.
__global__ void colsum_kernel(const float* __restrict__ ekp, float* __restrict__ zpart)
{
    const int t = threadIdx.x, b = blockIdx.x;
    float s = 0.0f;
    const int base = b * 256;
    for (int r = 0; r < 256; ++r) s += ekp[(base + r) * DIM + t];
    zpart[b * DIM + t] = s;
}

__global__ void zreduce_kernel(const float* __restrict__ zpart, float* __restrict__ z)
{
    const int t = threadIdx.x;
    float s = 0.0f;
    for (int b = 0; b < 32; ++b) s += zpart[b * DIM + t];
    z[t] = s;
}

__global__ void sreduce_kernel(const float* __restrict__ Spart, float* __restrict__ S)
{
    const int i = blockIdx.x * blockDim.x + threadIdx.x;
    float s = 0.0f;
    for (int p = 0; p < 16; ++p) s += Spart[(size_t)p * DIM * DIM + i];
    S[i] = s;
}

// num[row,:] /= (eqp[row,:].z + 1e-8), exact f32 denominator. Block per row.
__global__ void attnscale_kernel(const float* __restrict__ eqp, const float* __restrict__ z,
                                 float* __restrict__ num)
{
    __shared__ float red[256];
    const int row = blockIdx.x, t = threadIdx.x;
    red[t] = eqp[row * DIM + t] * z[t];
    __syncthreads();
    for (int s = 128; s > 0; s >>= 1) {
        if (t < s) red[t] += red[t + s];
        __syncthreads();
    }
    const float den = red[0] + 1e-8f;
    num[row * DIM + t] = num[row * DIM + t] / den;
}

// out = LayerNorm(a + b) * g + beta. Block per row, D = 256 = blockDim.
__global__ void ln_kernel(const float* __restrict__ a, const float* __restrict__ b,
                          const float* __restrict__ g, const float* __restrict__ beta,
                          float* __restrict__ out)
{
    __shared__ float red[256];
    const int row = blockIdx.x, t = threadIdx.x;
    const float s = a[row * DIM + t] + b[row * DIM + t];
    red[t] = s;
    __syncthreads();
    for (int k = 128; k > 0; k >>= 1) {
        if (t < k) red[t] += red[t + k];
        __syncthreads();
    }
    const float mean = red[0] * (1.0f / DIM);
    __syncthreads();
    const float d = s - mean;
    red[t] = d * d;
    __syncthreads();
    for (int k = 128; k > 0; k >>= 1) {
        if (t < k) red[t] += red[t + k];
        __syncthreads();
    }
    const float var = red[0] * (1.0f / DIM);
    out[row * DIM + t] = d * rsqrtf(var + 1e-5f) * g[t] + beta[t];
}

// ---------------------------------------------------------------------------
extern "C" void kernel_launch(void* const* d_in, const int* in_sizes, int n_in,
                              void* d_out, int out_size, void* d_ws, size_t ws_size,
                              hipStream_t stream)
{
    const float* x   = (const float*)d_in[0];
    const float* Wq  = (const float*)d_in[1];
    const float* Wk  = (const float*)d_in[2];
    const float* Wv  = (const float*)d_in[3];
    const float* Wo  = (const float*)d_in[4];
    const float* W1  = (const float*)d_in[5];
    const float* b1  = (const float*)d_in[6];
    const float* W2  = (const float*)d_in[7];
    const float* b2  = (const float*)d_in[8];
    const float* g1  = (const float*)d_in[9];
    const float* be1 = (const float*)d_in[10];
    const float* g2  = (const float*)d_in[11];
    const float* be2 = (const float*)d_in[12];
    const float* R   = (const float*)d_in[13];
    float* out = (float*)d_out;

    const size_t ND = (size_t)NROWS * DIM;
    float* ws    = (float*)d_ws;
    float* bQ    = ws;              // q  -> eqp -> ff1
    float* bK    = bQ + ND;         // k  -> ekp -> ff2
    float* bV    = bK + ND;         // v  -> h
    float* bQP   = bV + ND;         // qp -> num -> attn
    float* bKP   = bQP + ND;        // kp -> ao
    float* Spart = bKP + ND;        // 16 * 256*256
    float* S     = Spart + 16 * DIM * DIM;
    float* zpart = S + DIM * DIM;   // 32 * 256
    float* z     = zpart + 32 * DIM;

    dim3 blk(256);
    dim3 gBig(DIM / 128, NROWS / 128);   // (2, 64)

    // pre-exp chain: exact f32 WMMA
    gemm_f32_kernel<<<gBig, blk, 0, stream>>>(x, Wq, bQ, NROWS, DIM, DIM);
    gemm_f32_kernel<<<gBig, blk, 0, stream>>>(x, Wk, bK, NROWS, DIM, DIM);
    gemm_f32_kernel<<<gBig, blk, 0, stream>>>(x, Wv, bV, NROWS, DIM, DIM);
    gemm_f32_kernel<<<gBig, blk, 0, stream>>>(bQ, R, bQP, NROWS, DIM, DIM);
    gemm_f32_kernel<<<gBig, blk, 0, stream>>>(bK, R, bKP, NROWS, DIM, DIM);

    rowexp_kernel<<<NROWS, blk, 0, stream>>>(bQP, bQ);    // eqp
    rowexp_kernel<<<NROWS, blk, 0, stream>>>(bKP, bK);    // ekp

    colsum_kernel<<<32, blk, 0, stream>>>(bK, zpart);
    zreduce_kernel<<<1, blk, 0, stream>>>(zpart, z);

    splitk_bf16_kernel<<<dim3(2, 2, 16), blk, 0, stream>>>(bK, bV, Spart); // S = ekp^T v
    sreduce_kernel<<<DIM, blk, 0, stream>>>(Spart, S);

    gemm_bf16_kernel<<<gBig, blk, 0, stream>>>(bQ, S, nullptr, bQP,
                                               NROWS, DIM, DIM, 0);        // num
    attnscale_kernel<<<NROWS, blk, 0, stream>>>(bQ, z, bQP);               // attn@v
    gemm_bf16_kernel<<<gBig, blk, 0, stream>>>(bQP, Wo, nullptr, bKP,
                                               NROWS, DIM, DIM, 0);        // ao

    ln_kernel<<<NROWS, blk, 0, stream>>>(x, bKP, g1, be1, bV);             // h

    gemm_bf16_kernel<<<gBig, blk, 0, stream>>>(bV, W1, b1, bQ,
                                               NROWS, DIM, DIM, 2);        // relu(h W1 + b1)
    gemm_bf16_kernel<<<gBig, blk, 0, stream>>>(bQ, W2, b2, bK,
                                               NROWS, DIM, DIM, 1);        // ff2
    ln_kernel<<<NROWS, blk, 0, stream>>>(bV, bK, g2, be2, out);            // output
}
